// Mamba_21818433863940
// MI455X (gfx1250) — compile-verified, matrix operand order checked by reading, not run on hardware
//
#include <hip/hip_runtime.h>
#include <hip/hip_bf16.h>

// ---------------------------------------------------------------------------
// Mamba forward for MI455X (gfx1250). All GEMMs via v_wmma_f32_16x16x32_bf16.
// Branch-free inner loops: conv handled by zero-padded A rows, K=48 padded to 64.
// ---------------------------------------------------------------------------

typedef __bf16 v16bf __attribute__((ext_vector_type(16)));
typedef __bf16 v8bf  __attribute__((ext_vector_type(8)));
typedef float  v8f   __attribute__((ext_vector_type(8)));

#define SEQ 1024
#define DM  768
#define DIN 1536
#define DST 16
#define DTR 48
#define DTRP 64      // DTR padded to a multiple of 64 for the WMMA K-loop
#define VOC 32000
#define NL  2

// ---------------------------------------------------------------------------
// elementwise / utility kernels
// ---------------------------------------------------------------------------

__global__ void k_cvt_f32_bf16(const float* __restrict__ s, __bf16* __restrict__ d, long n) {
    long i = (long)blockIdx.x * blockDim.x + threadIdx.x;
    long stride = (long)gridDim.x * blockDim.x;
    for (; i < n; i += stride) d[i] = (__bf16)s[i];
}

// convert [rows x srcC] f32 -> [rows x dstC] bf16, zero-filling columns >= srcC
__global__ void k_cvt_pad(const float* __restrict__ s, __bf16* __restrict__ d,
                          long rows, int srcC, int dstC) {
    long n = rows * dstC;
    long i = (long)blockIdx.x * blockDim.x + threadIdx.x;
    long stride = (long)gridDim.x * blockDim.x;
    for (; i < n; i += stride) {
        long r = i / dstC; int c = (int)(i - r * dstC);
        d[i] = (c < srcC) ? (__bf16)s[r * srcC + c] : (__bf16)0.0f;
    }
}

__global__ void k_zero_bf16(__bf16* __restrict__ d, long n) {
    long i = (long)blockIdx.x * blockDim.x + threadIdx.x;
    long stride = (long)gridDim.x * blockDim.x;
    for (; i < n; i += stride) d[i] = (__bf16)0.0f;
}

// W_conv slice [O][C][4] -> bf16 [O][C] for tap j (stride-4 gather)
__global__ void k_pack_tap(const float* __restrict__ w, __bf16* __restrict__ d,
                           long n /* O*C */, int j) {
    long i = (long)blockIdx.x * blockDim.x + threadIdx.x;
    long stride = (long)gridDim.x * blockDim.x;
    for (; i < n; i += stride) d[i] = (__bf16)w[i * 4 + j];
}

__global__ void k_embed(const int* __restrict__ ids, const float* __restrict__ emb,
                        float* __restrict__ e) {
    int t = blockIdx.x;
    long id = (long)ids[t];
    for (int i = threadIdx.x; i < DM; i += blockDim.x)
        e[(long)t * DM + i] = emb[id * DM + i];
}

__global__ void k_rmsnorm_bf16(const float* __restrict__ x, const float* __restrict__ w,
                               __bf16* __restrict__ out, int D) {
    int row = blockIdx.x;
    const float* xr = x + (long)row * D;
    __shared__ float red[256];
    float s = 0.f;
    for (int i = threadIdx.x; i < D; i += blockDim.x) { float v = xr[i]; s += v * v; }
    red[threadIdx.x] = s; __syncthreads();
    for (int off = 128; off > 0; off >>= 1) {
        if (threadIdx.x < off) red[threadIdx.x] += red[threadIdx.x + off];
        __syncthreads();
    }
    float rinv = rsqrtf(red[0] / (float)D + 1e-5f);
    for (int i = threadIdx.x; i < D; i += blockDim.x)
        out[(long)row * D + i] = (__bf16)(xr[i] * rinv * w[i]);
}

__global__ void k_silu_dual(const float* __restrict__ x, float* __restrict__ uf,
                            __bf16* __restrict__ ub, long n) {
    long i = (long)blockIdx.x * blockDim.x + threadIdx.x;
    long stride = (long)gridDim.x * blockDim.x;
    for (; i < n; i += stride) {
        float v = x[i];
        float s = v / (1.f + __expf(-v));
        uf[i] = s; ub[i] = (__bf16)s;
    }
}

__global__ void k_softplus_inplace(float* __restrict__ x, long n) {
    long i = (long)blockIdx.x * blockDim.x + threadIdx.x;
    long stride = (long)gridDim.x * blockDim.x;
    for (; i < n; i += stride) {
        float v = x[i];
        x[i] = (v > 20.f) ? v : log1pf(__expf(v));
    }
}

__global__ void k_gate(const float* __restrict__ y, const float* __restrict__ res,
                       __bf16* __restrict__ out, long n) {
    long i = (long)blockIdx.x * blockDim.x + threadIdx.x;
    long stride = (long)gridDim.x * blockDim.x;
    for (; i < n; i += stride) {
        float r = res[i];
        float s = r / (1.f + __expf(-r));
        out[i] = (__bf16)(y[i] * s);
    }
}

// selective scan: one thread per channel d, 16 independent state recurrences
__global__ void k_scan(const float* __restrict__ delta, const float* __restrict__ u,
                       const float* __restrict__ Bm, const float* __restrict__ Cm,
                       const float* __restrict__ A_log, const float* __restrict__ Dp,
                       float* __restrict__ y) {
    int d = blockIdx.x * blockDim.x + threadIdx.x;
    if (d >= DIN) return;
    float a[DST], s[DST];
#pragma unroll
    for (int n = 0; n < DST; ++n) { a[n] = -__expf(A_log[(long)d * DST + n]); s[n] = 0.f; }
    float dp = Dp[d];
    for (int t = 0; t < SEQ; ++t) {
        float dl = delta[(long)t * DIN + d];
        float uu = u[(long)t * DIN + d];
        float du = dl * uu;
        const float* bt = Bm + (long)t * DST;
        const float* ct = Cm + (long)t * DST;
        float acc = 0.f;
#pragma unroll
        for (int n = 0; n < DST; ++n) {
            s[n] = __expf(dl * a[n]) * s[n] + du * bt[n];
            acc += s[n] * ct[n];
        }
        y[(long)t * DIN + d] = acc + uu * dp;
    }
}

// ---------------------------------------------------------------------------
// WMMA fragment load (branch-free): row = lane%16, K runs {k0..k0+7} and
// {k0+16..k0+23}, k0 = kc + (lane>>4)*8. Two 16B vector loads.
// ---------------------------------------------------------------------------

__device__ __forceinline__ v16bf ld_frag(const __bf16* __restrict__ prow, int k0) {
    v8bf lo = *(const v8bf*)(prow + k0);
    v8bf hi = *(const v8bf*)(prow + k0 + 16);
    v16bf f;
#pragma unroll
    for (int e = 0; e < 8; ++e) { f[e] = lo[e]; f[e + 8] = hi[e]; }
    return f;
}

__device__ __forceinline__ v8f wmma_bf16(v16bf a, v16bf b, v8f c) {
    return __builtin_amdgcn_wmma_f32_16x16x32_bf16(false, a, false, b, (short)0, c,
                                                   false, false);
}

// ---------------------------------------------------------------------------
// 32x32-per-wave NT GEMM: C[m,n] = sum_k A[m,k]*B[n,k] (+C if acc)(+bias[n])
// Requires M%32==0 (A rows >= M+rows consumed), N%32==0, K%64==0.
// ---------------------------------------------------------------------------

__global__ void k_gemm32(const __bf16* __restrict__ A, const __bf16* __restrict__ B,
                         float* __restrict__ C, const float* __restrict__ bias,
                         int M, int N, int K, int lda, int ldb, int ldc,
                         int accumulate) {
    int wid  = (blockIdx.x * blockDim.x + threadIdx.x) >> 5;
    int lane = threadIdx.x & 31;
    int Mt = M >> 5, Nt = N >> 5;
    if (wid >= Mt * Nt) return;              // wave-uniform exit
    int tm = wid % Mt, tn = wid / Mt;        // consecutive waves share B strip
    int m0 = tm << 5, n0 = tn << 5;

    int row   = lane & 15;
    int kbase = (lane >> 4) << 3;            // 0 or 8
    int mhi   = (lane >= 16) ? 8 : 0;
    int nlo   = lane & 15;

    const __bf16* pa0 = A + (long)(m0 + row) * lda;
    const __bf16* pa1 = pa0 + (long)16 * lda;
    const __bf16* pb0 = B + (long)(n0 + row) * ldb;
    const __bf16* pb1 = pb0 + (long)16 * ldb;

    v8f c00, c01, c10, c11;
    if (accumulate) {
#pragma unroll
        for (int r = 0; r < 8; ++r) {
            long rm = (long)(m0 + r + mhi) * ldc;
            c00[r] = C[rm + n0 + nlo];
            c01[r] = C[rm + n0 + 16 + nlo];
            c10[r] = C[rm + (long)16 * ldc + n0 + nlo];
            c11[r] = C[rm + (long)16 * ldc + n0 + 16 + nlo];
        }
    } else {
#pragma unroll
        for (int r = 0; r < 8; ++r) { c00[r] = c01[r] = c10[r] = c11[r] = 0.f; }
    }

#pragma unroll 2
    for (int kc = 0; kc < K; kc += 32) {
        int k0 = kc + kbase;
        v16bf a0 = ld_frag(pa0, k0);
        v16bf a1 = ld_frag(pa1, k0);
        v16bf b0 = ld_frag(pb0, k0);
        v16bf b1 = ld_frag(pb1, k0);
        c00 = wmma_bf16(a0, b0, c00);
        c01 = wmma_bf16(a0, b1, c01);
        c10 = wmma_bf16(a1, b0, c10);
        c11 = wmma_bf16(a1, b1, c11);
    }

    float bv0 = bias ? bias[n0 + nlo] : 0.f;
    float bv1 = bias ? bias[n0 + 16 + nlo] : 0.f;
#pragma unroll
    for (int r = 0; r < 8; ++r) {
        long rm = (long)(m0 + r + mhi) * ldc;
        C[rm + n0 + nlo]                      = c00[r] + bv0;
        C[rm + n0 + 16 + nlo]                 = c01[r] + bv1;
        C[rm + (long)16 * ldc + n0 + nlo]     = c10[r] + bv0;
        C[rm + (long)16 * ldc + n0 + 16 + nlo]= c11[r] + bv1;
    }
}

// ---------------------------------------------------------------------------
// 16x16-per-wave NT GEMM for the skinny projections (N=16/48). K%64==0.
// ---------------------------------------------------------------------------

__global__ void k_gemm16(const __bf16* __restrict__ A, const __bf16* __restrict__ B,
                         float* __restrict__ C, const float* __restrict__ bias,
                         int M, int N, int K, int lda, int ldb, int ldc) {
    int wid  = (blockIdx.x * blockDim.x + threadIdx.x) >> 5;
    int lane = threadIdx.x & 31;
    int Mt = M >> 4, Nt = N >> 4;
    if (wid >= Mt * Nt) return;
    int tm = wid % Mt, tn = wid / Mt;
    int m0 = tm << 4, n0 = tn << 4;

    int row   = lane & 15;
    int kbase = (lane >> 4) << 3;
    int mhi   = (lane >= 16) ? 8 : 0;
    int ncol  = n0 + (lane & 15);

    const __bf16* pa = A + (long)(m0 + row) * lda;
    const __bf16* pb = B + (long)(n0 + row) * ldb;

    v8f c;
#pragma unroll
    for (int r = 0; r < 8; ++r) c[r] = 0.f;

#pragma unroll 2
    for (int kc = 0; kc < K; kc += 32) {
        int k0 = kc + kbase;
        c = wmma_bf16(ld_frag(pa, k0), ld_frag(pb, k0), c);
    }

    float bv = bias ? bias[ncol] : 0.f;
#pragma unroll
    for (int r = 0; r < 8; ++r)
        C[(long)(m0 + r + mhi) * ldc + ncol] = c[r] + bv;
}

// ---------------------------------------------------------------------------
// host orchestration
// ---------------------------------------------------------------------------

extern "C" void kernel_launch(void* const* d_in, const int* in_sizes, int n_in,
                              void* d_out, int out_size, void* d_ws, size_t ws_size,
                              hipStream_t stream) {
    const int*   ids      = (const int*)  d_in[0];
    const float* emb      = (const float*)d_in[1];
    const float* W_res    = (const float*)d_in[2];
    const float* b_res    = (const float*)d_in[3];
    const float* W_state  = (const float*)d_in[4];
    const float* b_state  = (const float*)d_in[5];
    const float* W_conv   = (const float*)d_in[6];
    const float* W_dt     = (const float*)d_in[7];
    const float* b_dt     = (const float*)d_in[8];
    const float* W_B      = (const float*)d_in[9];
    const float* b_B      = (const float*)d_in[10];
    const float* W_C      = (const float*)d_in[11];
    const float* b_C      = (const float*)d_in[12];
    const float* W_dtp    = (const float*)d_in[13];
    const float* b_dtp    = (const float*)d_in[14];
    const float* A_log    = (const float*)d_in[15];
    const float* D_param  = (const float*)d_in[16];
    const float* W_out    = (const float*)d_in[17];
    const float* b_out    = (const float*)d_in[18];
    const float* norm_w   = (const float*)d_in[19];
    const float* fnorm_w  = (const float*)d_in[20];
    float* logits = (float*)d_out;

    // workspace bump allocator
    char* ws = (char*)d_ws;
    size_t off = 0;
    auto alloc = [&](size_t bytes) -> void* {
        void* p = ws + off;
        off += (bytes + 255) & ~(size_t)255;
        return p;
    };
    __bf16* emb_bf   = (__bf16*)alloc((size_t)VOC * DM * 2);
    float*  e        = (float*) alloc((size_t)SEQ * DM * 4);
    __bf16* xn_bf    = (__bf16*)alloc((size_t)SEQ * DM * 2);
    float*  res      = (float*) alloc((size_t)SEQ * DIN * 4);
    float*  xbuf     = (float*) alloc((size_t)SEQ * DIN * 4);
    __bf16* x_pad    = (__bf16*)alloc((size_t)(SEQ + 3) * DIN * 2); // 3 zero rows front
    float*  xc       = (float*) alloc((size_t)SEQ * DIN * 4);
    float*  u        = (float*) alloc((size_t)SEQ * DIN * 4);
    __bf16* u_bf     = (__bf16*)alloc((size_t)SEQ * DIN * 2);
    float*  delta    = (float*) alloc((size_t)SEQ * DIN * 4);
    float*  tmp48    = (float*) alloc((size_t)SEQ * DTR * 4);
    __bf16* tmp48_bf = (__bf16*)alloc((size_t)SEQ * DTRP * 2);     // K padded to 64
    float*  Bmat     = (float*) alloc((size_t)SEQ * DST * 4);
    float*  Cmat     = (float*) alloc((size_t)SEQ * DST * 4);
    float*  ybuf     = (float*) alloc((size_t)SEQ * DIN * 4);
    __bf16* y_bf     = (__bf16*)alloc((size_t)SEQ * DIN * 2);
    __bf16* Wbuf     = (__bf16*)alloc((size_t)DIN * DIN * 2);
    (void)ws_size; (void)in_sizes; (void)n_in; (void)out_size;

    auto cvt = [&](const float* s, __bf16* d, long n) {
        k_cvt_f32_bf16<<<dim3(2048), dim3(256), 0, stream>>>(s, d, n);
    };
    auto gemm32 = [&](const __bf16* A, const __bf16* B, float* C, const float* bias,
                      int M, int N, int K, int lda, int ldb, int ldc, int acc) {
        long tiles  = (long)(M / 32) * (N / 32);
        long blocks = (tiles + 7) / 8;                  // 8 waves / 256-thread block
        k_gemm32<<<dim3((unsigned)blocks), dim3(256), 0, stream>>>(
            A, B, C, bias, M, N, K, lda, ldb, ldc, acc);
    };
    auto gemm16 = [&](const __bf16* A, const __bf16* B, float* C, const float* bias,
                      int M, int N, int K, int lda, int ldb, int ldc) {
        long tiles  = (long)(M / 16) * (N / 16);
        long blocks = (tiles + 7) / 8;
        k_gemm16<<<dim3((unsigned)blocks), dim3(256), 0, stream>>>(
            A, B, C, bias, M, N, K, lda, ldb, ldc);
    };

    // embedding table -> bf16 (B of the tied logits GEMM)
    cvt(emb, emb_bf, (long)VOC * DM);
    // token gather
    k_embed<<<dim3(SEQ), dim3(256), 0, stream>>>(ids, emb, e);
    // zero the 3 leading pad rows of the conv input once
    k_zero_bf16<<<dim3(8), dim3(256), 0, stream>>>(x_pad, (long)3 * DIN);

    for (int i = 0; i < NL; ++i) {
        const long wpo = (long)i * DIN * DM;     // W_res / W_state slice
        // xn = rmsnorm(e) (bf16)
        k_rmsnorm_bf16<<<dim3(SEQ), dim3(256), 0, stream>>>(e, norm_w + (long)i * DM, xn_bf, DM);

        // res = xn @ W_res^T + b_res
        cvt(W_res + wpo, Wbuf, (long)DIN * DM);
        gemm32(xn_bf, Wbuf, res, b_res + (long)i * DIN, SEQ, DIN, DM, DM, DM, DIN, 0);
        // x = xn @ W_state^T + b_state
        cvt(W_state + wpo, Wbuf, (long)DIN * DM);
        gemm32(xn_bf, Wbuf, xbuf, b_state + (long)i * DIN, SEQ, DIN, DM, DM, DM, DIN, 0);
        cvt(xbuf, x_pad + (long)3 * DIN, (long)SEQ * DIN);

        // causal full conv as 4 shifted accumulating GEMMs (zero-padded rows)
        const float* wc = W_conv + (long)i * DIN * DIN * 4;
        for (int j = 0; j < 4; ++j) {
            k_pack_tap<<<dim3(2048), dim3(256), 0, stream>>>(wc, Wbuf, (long)DIN * DIN, j);
            gemm32(x_pad + (long)j * DIN, Wbuf, xc, nullptr,
                   SEQ, DIN, DIN, DIN, DIN, DIN, (j > 0) ? 1 : 0);
        }

        // u = silu(xc) (f32 + bf16)
        k_silu_dual<<<dim3(2048), dim3(256), 0, stream>>>(xc, u, u_bf, (long)SEQ * DIN);

        // delta = softplus((u @ W_dt^T + b_dt) @ W_dtp^T + b_dtp)
        cvt(W_dt + (long)i * DTR * DIN, Wbuf, (long)DTR * DIN);
        gemm16(u_bf, Wbuf, tmp48, b_dt + (long)i * DTR, SEQ, DTR, DIN, DIN, DIN, DTR);
        k_cvt_pad<<<dim3(512), dim3(256), 0, stream>>>(tmp48, tmp48_bf, SEQ, DTR, DTRP);
        k_cvt_pad<<<dim3(512), dim3(256), 0, stream>>>(W_dtp + (long)i * DIN * DTR, Wbuf,
                                                       DIN, DTR, DTRP);
        gemm32(tmp48_bf, Wbuf, delta, b_dtp + (long)i * DIN, SEQ, DIN, DTRP, DTRP, DTRP, DIN, 0);
        k_softplus_inplace<<<dim3(2048), dim3(256), 0, stream>>>(delta, (long)SEQ * DIN);

        // B, C projections (N=16)
        cvt(W_B + (long)i * DST * DIN, Wbuf, (long)DST * DIN);
        gemm16(u_bf, Wbuf, Bmat, b_B + (long)i * DST, SEQ, DST, DIN, DIN, DIN, DST);
        cvt(W_C + (long)i * DST * DIN, Wbuf, (long)DST * DIN);
        gemm16(u_bf, Wbuf, Cmat, b_C + (long)i * DST, SEQ, DST, DIN, DIN, DIN, DST);

        // selective scan (sequential over time)
        k_scan<<<dim3((DIN + 255) / 256), dim3(256), 0, stream>>>(
            delta, u, Bmat, Cmat, A_log + (long)i * DIN * DST, D_param + (long)i * DIN, ybuf);

        // y = y * silu(res) (bf16)
        k_gate<<<dim3(2048), dim3(256), 0, stream>>>(ybuf, res, y_bf, (long)SEQ * DIN);

        // e += y @ W_out^T + b_out   (accumulate into residual in place)
        cvt(W_out + (long)i * DM * DIN, Wbuf, (long)DM * DIN);
        gemm32(y_bf, Wbuf, e, b_out + (long)i * DM, SEQ, DM, DIN, DIN, DIN, DM, 1);
    }

    // final norm + tied logits head
    k_rmsnorm_bf16<<<dim3(SEQ), dim3(256), 0, stream>>>(e, fnorm_w, xn_bf, DM);
    gemm32(xn_bf, emb_bf, logits, nullptr, SEQ, VOC, DM, DM, DM, VOC, 0);
}
